// StickyHDPHMMVI_6674379178021
// MI455X (gfx1250) — matrix operand order repaired and checked by reading, not compiled, not run on hardware
//
#include <hip/hip_runtime.h>
#include <hip/hip_bf16.h>

typedef __attribute__((ext_vector_type(16))) __bf16 v16bf;
typedef __attribute__((ext_vector_type(8)))  float  v8f;

#define BB 8
#define TT 4096
#define KK 16
#define DD 64
#define RR 8
#define TB 16           // timesteps per emission tile
#define NCOL ((RR + 1) * TB)   // 144 columns of V
#define VPAD 80         // padded row length (bf16) -> 160B, keeps 16/32B alignment

// ---- workspace layout (byte offsets) ----
#define WS_UFRAG   0                                  // 16*4096 bf16 = 131072 B
#define WS_CONST   131072                             // 16 f32
#define WS_C2      (WS_CONST + 64)                    // 16 f32
#define WS_W       (WS_C2 + 64)                       // 16*64 f32
#define WS_GDIAG   (WS_W + 4096)                      // 16*64 f32
#define WS_LOGB    (WS_GDIAG + 4096)                  // B*T*K f32 = 2 MB
#define WS_LALPHA  (WS_LOGB   + BB * TT * KK * 4)     // 2 MB
#define WS_LBETA   (WS_LALPHA + BB * TT * KK * 4)     // 2 MB

__device__ __forceinline__ float digammaf_dev(float x) {
  float r = 0.f;
  while (x < 6.f) { r -= 1.f / x; x += 1.f; }
  float inv  = 1.f / x;
  float inv2 = inv * inv;
  float s = __logf(x) - 0.5f * inv
          - inv2 * (0.083333333333f - inv2 * (0.0083333333333f - inv2 * 0.0039682539683f));
  return s + r;
}

// 32-byte LDS fragment load as two aligned 16B chunks -> 2x ds_load_b128
__device__ __forceinline__ v16bf lds_frag32(const __bf16* p) {
  const uint4* q = (const uint4*)__builtin_assume_aligned(p, 16);
  union { uint4 u[2]; v16bf v; } c;
  c.u[0] = q[0];
  c.u[1] = q[1];
  return c.v;
}

// ---------------------------------------------------------------------------
// Kernel 1: per-k Cholesky, U = sqrt(nu)*L^{-1}, const_k, w_k, ||w_k||^2,
// diag(E_Lam), and U written as bf16 in WMMA A-fragment lane order.
// ---------------------------------------------------------------------------
__global__ __launch_bounds__(64) void prep_kernel(
    const float* __restrict__ niw_mu, const float* __restrict__ niw_Psi,
    const float* __restrict__ niw_nu,
    __bf16* __restrict__ Ufrag, float* __restrict__ constk, float* __restrict__ c2k,
    float* __restrict__ wk, float* __restrict__ gdiag)
{
  __shared__ float P[DD][DD + 1];
  __shared__ float Uf[DD][DD + 1];
  __shared__ float red[DD];
  __shared__ float shmu[DD];
  __shared__ float stash[2];

  const int k   = blockIdx.x;
  const int tid = threadIdx.x;                // 0..63
  const float nu = niw_nu[k];

  for (int j = 0; j < DD; ++j)
    P[j][tid] = niw_Psi[(k * DD + j) * DD + tid];
  shmu[tid] = niw_mu[k * DD + tid];
  __syncthreads();

  // Cholesky (lower, in place); thread tid owns row tid
  for (int j = 0; j < DD; ++j) {
    if (tid == j) {
      float s = P[j][j];
      for (int m = 0; m < j; ++m) s -= P[j][m] * P[j][m];
      P[j][j] = sqrtf(fmaxf(s, 1e-20f));
    }
    __syncthreads();
    if (tid > j) {
      float s = P[tid][j];
      for (int m = 0; m < j; ++m) s -= P[tid][m] * P[j][m];
      P[tid][j] = s / P[j][j];
    }
    __syncthreads();
  }

  // U = sqrt(nu) * L^{-1}; thread tid owns COLUMN c (column-private, no races)
  {
    const int c = tid;
    const float snu = sqrtf(nu);
    for (int i = 0; i < DD; ++i) {
      float v;
      if (i < c)       v = 0.f;
      else if (i == c) v = 1.f / P[c][c];
      else {
        float s = 0.f;
        for (int m = c; m < i; ++m) s += P[i][m] * Uf[m][c];
        v = -s / P[i][i];
      }
      Uf[i][c] = v;
    }
    for (int i = 0; i < DD; ++i) Uf[i][c] *= snu;
  }
  __syncthreads();

  // logdet_Psi = 2 * sum log Ljj
  red[tid] = 2.f * __logf(P[tid][tid]);
  __syncthreads();
  for (int s = 32; s > 0; s >>= 1) { if (tid < s) red[tid] += red[tid + s]; __syncthreads(); }
  if (tid == 0) stash[0] = red[0];
  __syncthreads();
  // sum_i digamma((nu + 1 - i)/2), i = 1..64
  red[tid] = digammaf_dev((nu + 1.f - (float)(tid + 1)) * 0.5f);
  __syncthreads();
  for (int s = 32; s > 0; s >>= 1) { if (tid < s) red[tid] += red[tid + s]; __syncthreads(); }
  if (tid == 0) stash[1] = red[0];
  __syncthreads();

  // w = U * niw_mu (U lower triangular), c2 = ||w||^2
  float wi = 0.f;
  for (int m = 0; m <= tid; ++m) wi += Uf[tid][m] * shmu[m];
  wk[k * DD + tid] = wi;
  red[tid] = wi * wi;
  __syncthreads();
  for (int s = 32; s > 0; s >>= 1) { if (tid < s) red[tid] += red[tid + s]; __syncthreads(); }
  if (tid == 0) {
    c2k[k] = red[0];
    const float logdet   = stash[0];
    const float digsum   = stash[1];
    const float E_logdet = digsum + 64.f * 0.69314718056f - logdet;
    constk[k] = 0.5f * (E_logdet - 64.f * 1.83787706641f);   // log(2*pi)
  }

  // Gdiag[d] = sum_i U[i][d]^2  (diag of E_Lam = U^T U)
  float gd = 0.f;
  for (int i = tid; i < DD; ++i) gd += Uf[i][tid] * Uf[i][tid];
  gdiag[k * DD + tid] = gd;

  // Emit U as bf16 A fragments in this file's contiguous-k order (B loads match):
  // lane<16 holds row m, K = {0..15}+32*ki; lane>=16 holds K = {16..31}+32*ki
  for (int f = tid; f < DD * DD; f += 64) {
    const int e    = f & 15;
    const int lane = (f >> 4) & 31;
    const int mi   = (f >> 9) & 3;
    const int ki   = (f >> 11) & 1;
    const int m    = lane & 15;
    const int base = (lane >= 16) ? 16 : 0;
    const int kcol = ki * 32 + base + e;
    const int row  = mi * 16 + m;
    Ufrag[k * (DD * DD) + f] = (__bf16)Uf[row][kcol];
  }
}

// ---------------------------------------------------------------------------
// Kernel 2: emission logB via WMMA. One block = 16 timesteps of one batch.
// 4 waves, each owning 4 k's. A fragments hoisted per-k (8 frags in VGPRs),
// B fragments hoisted per column-group; 4 live accumulators so 4 consecutive
// WMMAs share one B operand.
// ---------------------------------------------------------------------------
__global__ __launch_bounds__(128) void emis_kernel(
    const float* __restrict__ mu_t, const float* __restrict__ var_t,
    const float* __restrict__ F_t,
    const __bf16* __restrict__ Ufrag, const float* __restrict__ constk,
    const float* __restrict__ c2k, const float* __restrict__ wk,
    const float* __restrict__ gdiag, float* __restrict__ logB)
{
  __shared__ __align__(32) __bf16 Vs[NCOL][VPAD];  // V column-major: [col][d]
  __shared__ float vars[TB][DD + 1];
  __shared__ float wsh[KK][DD];
  __shared__ float gdsh[KK][DD];
  __shared__ float csh[KK], c2sh[KK];

  const int b   = blockIdx.y;
  const int t0  = blockIdx.x * TB;
  const int tid = threadIdx.x;
  const int lane = tid & 31;
  const int wave = tid >> 5;

  // stage V = [mu | F columns] (bf16) + vars + per-k constants
  for (int i = tid; i < TB * DD; i += 128) {
    const int t = i >> 6, d = i & 63;
    const size_t g = ((size_t)(b * TT + t0 + t)) * DD + d;
    Vs[t][d]   = (__bf16)mu_t[g];     // columns 0..15 = mu for t=0..15
    vars[t][d] = var_t[g];
  }
  for (int i = tid; i < KK * DD; i += 128) {
    wsh[i >> 6][i & 63]  = wk[i];
    gdsh[i >> 6][i & 63] = gdiag[i];
  }
  if (tid < KK) { csh[tid] = constk[tid]; c2sh[tid] = c2k[tid]; }
  for (int i = tid; i < TB * DD * RR; i += 128) {
    const int r = i & 7, d = (i >> 3) & 63, t = i >> 9;
    const size_t g = (((size_t)(b * TT + t0 + t)) * DD + d) * RR + r;
    Vs[(1 + r) * TB + t][d] = (__bf16)F_t[g];    // column group j=1+r, col index t
  }
  __syncthreads();

  const int  tcol = lane & 15;
  const bool hihalf = lane >= 16;
  const int  boff = hihalf ? 16 : 0;   // element offset into a column for this half

  for (int kk = 0; kk < 4; ++kk) {
    const int k = wave * 4 + kk;
    const __bf16* Ub = Ufrag + k * (DD * DD);

    // Hoist all 8 A fragments for this k: index i = ki*4 + mi
    v16bf afr[8];
    #pragma unroll
    for (int i = 0; i < 8; ++i)
      afr[i] = *(const v16bf*)(Ub + ((size_t)(i * 32 + lane)) * 16);

    float qsum = 0.f, dsum = 0.f;

    for (int ci = 0; ci < 9; ++ci) {         // column groups: 0 = mu, 1..8 = F cols
      const __bf16* col = &Vs[ci * TB + tcol][0];
      v16bf b0 = lds_frag32(col + boff);        // K = 0..31 half for this lane
      v16bf b1 = lds_frag32(col + 32 + boff);   // K = 32..63 half

      v8f acc[4];
      #pragma unroll
      for (int mi = 0; mi < 4; ++mi) acc[mi] = (v8f){};
      #pragma unroll
      for (int mi = 0; mi < 4; ++mi)
        acc[mi] = __builtin_amdgcn_wmma_f32_16x16x32_bf16(false, afr[mi], false, b0,
                                                          (short)0, acc[mi], false, false);
      #pragma unroll
      for (int mi = 0; mi < 4; ++mi)
        acc[mi] = __builtin_amdgcn_wmma_f32_16x16x32_bf16(false, afr[4 + mi], false, b1,
                                                          (short)0, acc[mi], false, false);

      #pragma unroll
      for (int mi = 0; mi < 4; ++mi) {
        float p = 0.f;
        #pragma unroll
        for (int v = 0; v < 8; ++v) p += acc[mi][v] * acc[mi][v];  // ||U x||^2 partial
        qsum += p;
      }
      if (ci == 0) {                          // w_k . (U mu) partial for mean term
        #pragma unroll
        for (int mi = 0; mi < 4; ++mi) {
          float dp = 0.f;
          #pragma unroll
          for (int v = 0; v < 8; ++v)
            dp += wsh[k][mi * 16 + (hihalf ? 8 + v : v)] * acc[mi][v];
          dsum += dp;
        }
      }
    }
    qsum += __shfl_xor(qsum, 16, 32);         // combine the two half-wave row-sets
    dsum += __shfl_xor(dsum, 16, 32);

    if (!hihalf) {
      float trd = 0.f;
      for (int d = 0; d < DD; ++d) trd += gdsh[k][d] * vars[tcol][d];
      const float q = qsum + c2sh[k] - 2.f * dsum + trd;
      logB[((size_t)(b * TT + t0 + tcol)) * KK + k] = csh[k] - 0.5f * q;
    }
  }
}

// ---------------------------------------------------------------------------
// Kernel 3: log-domain forward / backward HMM scan. One wave per (b, dir).
// Whole logB sequence (256 KB) staged in LDS (CDNA5: 320 KB per workgroup).
// Scaled-linear recurrence: one log + one exp per step on the critical path.
// ---------------------------------------------------------------------------
__global__ __launch_bounds__(32) void fb_kernel(
    const float* __restrict__ logB, const float* __restrict__ log_pi,
    const float* __restrict__ ElogA,
    float* __restrict__ log_alpha, float* __restrict__ log_beta)
{
  extern __shared__ float smem[];
  float* lbs = smem;                 // TT*KK staged logB
  float* Ah  = smem + TT * KK;       // 16x16 transition (exp), transposed for bwd
  float* pS  = Ah + 256;             // normalized state vector

  const int  b    = blockIdx.x >> 1;
  const bool bwd  = blockIdx.x & 1;
  const int  lane = threadIdx.x;
  const int  k    = lane & 15;       // lanes 16..31 mirror 0..15
  const float* gB = logB + (size_t)b * TT * KK;

  {
    const float4* src = (const float4*)gB;
    float4* dst = (float4*)lbs;
    for (int i = lane; i < TT * KK / 4; i += 32) dst[i] = src[i];
  }
  for (int i = lane; i < 256; i += 32) {
    const int r = i >> 4, c = i & 15;
    Ah[r * 16 + c] = __expf(bwd ? ElogA[c * 16 + r] : ElogA[i]);
  }
  __syncthreads();

  float areg[16];
  #pragma unroll
  for (int j = 0; j < 16; ++j) areg[j] = Ah[j * 16 + k];  // Ah[j][k]

  if (!bwd) {
    float la = log_pi[k] + lbs[k];
    if (lane < 16) log_alpha[((size_t)b * TT) * KK + k] = la;
    float m = la;
    #pragma unroll
    for (int o = 8; o > 0; o >>= 1) m = fmaxf(m, __shfl_xor(m, o, 32));
    if (lane < 16) pS[k] = __expf(la - m);
    __syncthreads();
    for (int t = 1; t < TT; ++t) {
      float y = 0.f;
      #pragma unroll
      for (int j = 0; j < 16; ++j) y += pS[j] * areg[j];
      const float la2 = m + __logf(y) + lbs[t * 16 + k];
      if (lane < 16) log_alpha[((size_t)b * TT + t) * KK + k] = la2;
      float mn = la2;
      #pragma unroll
      for (int o = 8; o > 0; o >>= 1) mn = fmaxf(mn, __shfl_xor(mn, o, 32));
      __syncthreads();
      if (lane < 16) pS[k] = __expf(la2 - mn);
      m = mn;
      __syncthreads();
    }
  } else {
    if (lane < 16) log_beta[((size_t)b * TT + (TT - 1)) * KK + k] = 0.f;
    float v = lbs[(TT - 1) * 16 + k];        // lb(T-1)=0
    float m = v;
    #pragma unroll
    for (int o = 8; o > 0; o >>= 1) m = fmaxf(m, __shfl_xor(m, o, 32));
    if (lane < 16) pS[k] = __expf(v - m);
    __syncthreads();
    for (int t = TT - 2; t >= 0; --t) {
      float y = 0.f;
      #pragma unroll
      for (int j = 0; j < 16; ++j) y += pS[j] * areg[j];   // areg[j] = exp(ElogA[k][j])
      const float lb = m + __logf(y);
      if (lane < 16) log_beta[((size_t)b * TT + t) * KK + k] = lb;
      const float v2 = lb + lbs[t * 16 + k];
      float mn = v2;
      #pragma unroll
      for (int o = 8; o > 0; o >>= 1) mn = fmaxf(mn, __shfl_xor(mn, o, 32));
      __syncthreads();
      if (lane < 16) pS[k] = __expf(v2 - mn);
      m = mn;
      __syncthreads();
    }
  }
}

// ---------------------------------------------------------------------------
// Kernel 4: rhat = softmax_k(log_alpha + log_beta)
// ---------------------------------------------------------------------------
__global__ __launch_bounds__(256) void post_kernel(
    const float* __restrict__ la, const float* __restrict__ lb, float* __restrict__ out)
{
  const int i = blockIdx.x * blockDim.x + threadIdx.x;
  if (i >= BB * TT) return;
  float v[16];
  float m = -3.0e38f;
  #pragma unroll
  for (int kq = 0; kq < 16; ++kq) {
    v[kq] = la[(size_t)i * 16 + kq] + lb[(size_t)i * 16 + kq];
    m = fmaxf(m, v[kq]);
  }
  float s = 0.f;
  #pragma unroll
  for (int kq = 0; kq < 16; ++kq) { v[kq] = __expf(v[kq] - m); s += v[kq]; }
  const float inv = 1.f / s;
  #pragma unroll
  for (int kq = 0; kq < 16; ++kq) out[(size_t)i * 16 + kq] = v[kq] * inv;
}

extern "C" void kernel_launch(void* const* d_in, const int* in_sizes, int n_in,
                              void* d_out, int out_size, void* d_ws, size_t ws_size,
                              hipStream_t stream) {
  (void)in_sizes; (void)n_in; (void)out_size; (void)ws_size;
  const float* mu_t    = (const float*)d_in[0];
  const float* var_t   = (const float*)d_in[1];
  const float* F_t     = (const float*)d_in[2];
  const float* niw_mu  = (const float*)d_in[3];
  const float* niw_Psi = (const float*)d_in[4];
  const float* niw_nu  = (const float*)d_in[5];
  const float* log_pi  = (const float*)d_in[6];
  const float* ElogA   = (const float*)d_in[7];

  char* ws = (char*)d_ws;
  __bf16* Ufrag = (__bf16*)(ws + WS_UFRAG);
  float* constk = (float*)(ws + WS_CONST);
  float* c2k    = (float*)(ws + WS_C2);
  float* wk     = (float*)(ws + WS_W);
  float* gdiag  = (float*)(ws + WS_GDIAG);
  float* logB   = (float*)(ws + WS_LOGB);
  float* lalpha = (float*)(ws + WS_LALPHA);
  float* lbeta  = (float*)(ws + WS_LBETA);
  float* out    = (float*)d_out;

  prep_kernel<<<KK, 64, 0, stream>>>(niw_mu, niw_Psi, niw_nu,
                                     Ufrag, constk, c2k, wk, gdiag);
  emis_kernel<<<dim3(TT / TB, BB), 128, 0, stream>>>(mu_t, var_t, F_t, Ufrag,
                                                     constk, c2k, wk, gdiag, logB);
  const size_t fbsmem = (size_t)(TT * KK + 256 + 16) * sizeof(float);  // ~257 KB LDS
  fb_kernel<<<BB * 2, 32, fbsmem, stream>>>(logB, log_pi, ElogA, lalpha, lbeta);
  post_kernel<<<(BB * TT + 255) / 256, 256, 0, stream>>>(lalpha, lbeta, out);
}
